// GINModel_37056977830655
// MI455X (gfx1250) — matrix operand order repaired and checked by reading, not compiled, not run on hardware
//
#include <hip/hip_runtime.h>
#include <hip/hip_bf16.h>

#define Nn 100000
#define Ee 1600000
#define Dd 64
#define Hh 128
#define Ll 4
#define Gg 64

typedef __attribute__((ext_vector_type(16))) __bf16 v16bf;
typedef __attribute__((ext_vector_type(8)))  float  v8f;

// ---- WMMA bf16 16x16x32 per-lane element index maps (ISA 7.12.2) ----
// A (16xK tile, lane&15 = M row): VGPR j in 0..3 holds K=2j,2j+1 (lanes<16) /
// K=8+2j.. (lanes>=16); VGPR 4..7 hold K=16.. / K=24..
__device__ __forceinline__ int kofA(int t, int lane) {
  int j = t >> 1, p = t & 1;
  return ((j < 4) ? (2 * j) : (2 * j + 8)) + p + ((lane & 16) ? 8 : 0);
}
// B (Kx16 tile, lane&15 = N col): same K order as A: t<8 -> K=t (+8 for lanes>=16),
// t>=8 -> K=t+8 (+8 for lanes>=16)
__device__ __forceinline__ int kofB(int t, int lane) {
  return ((t < 8) ? t : (t + 8)) + ((lane & 16) ? 8 : 0);
}

__device__ __forceinline__ void lds_fence() {
  asm volatile("s_wait_dscnt 0x0" ::: "memory");
}

__device__ __forceinline__ v8f wmma_bf16(v16bf a, v16bf b, v8f c) {
  return __builtin_amdgcn_wmma_f32_16x16x32_bf16(false, a, false, b, (short)0, c,
                                                 false, false);
}

// ---------------- weight pre-pack into fragment-ready layout ----------------
// Packed frag f = ks*(Nw/16)+nt, element lane*16+t = B[32*ks+kofB(t,lane)][nt*16+(lane&15)]
__device__ __forceinline__ void pack_mat(const float* src, __bf16* dst, int Nw, int i) {
  int f = i >> 9;           // 512 elems per frag
  int r = i & 511;
  int lane = r >> 4, t = r & 15;
  int numNT = Nw >> 4;
  int ks = f / numNT, nt = f - ks * numNT;
  int k = 32 * ks + kofB(t, lane);
  int n = nt * 16 + (lane & 15);
  dst[i] = (__bf16)src[k * Nw + n];
}

__global__ __launch_bounds__(256) void pack_weights_kernel(
    const float* __restrict__ W_in, const float* __restrict__ W1,
    const float* __restrict__ W2, __bf16* __restrict__ wp) {
  int i = blockIdx.x * 256 + threadIdx.x;
  if (i < 64 * 64) pack_mat(W_in, wp, 64, i);
  if (i < Ll * 64 * 128) {
    int l = i / (64 * 128), j = i - l * (64 * 128);
    pack_mat(W1 + l * 64 * 128, wp + 4096 + l * 64 * 128, 128, j);
  }
  if (i < Ll * 128 * 64) {
    int l = i / (128 * 64), j = i - l * (128 * 64);
    pack_mat(W2 + l * 128 * 64, wp + 4096 + Ll * 8192 + l * 128 * 64, 64, j);
  }
}

// ---------------- h = x @ W_in + b_in  (WMMA, 16 rows per wave) ----------------
__global__ __launch_bounds__(256) void lin_in_kernel(
    const float* __restrict__ x, const __bf16* __restrict__ Wp,
    const float* __restrict__ b_in, float* __restrict__ h) {
  __shared__ float xs[8][16 * 64];
  int lane = threadIdx.x & 31;
  int wave = threadIdx.x >> 5;
  int tile = blockIdx.x * 8 + wave;
  if (tile * 16 >= Nn) return;
  long long row0 = (long long)tile * 16;
  float* T = &xs[wave][0];
  for (int i = lane; i < 16 * 64; i += 32) T[i] = x[row0 * 64 + i];
  lds_fence();

  v8f z8 = {0, 0, 0, 0, 0, 0, 0, 0};
  v8f acc[4];
#pragma unroll
  for (int i = 0; i < 4; ++i) acc[i] = z8;
  int m = lane & 15;
#pragma unroll
  for (int ks = 0; ks < 2; ++ks) {
    v16bf a;
#pragma unroll
    for (int t = 0; t < 16; ++t) a[t] = (__bf16)T[m * 64 + 32 * ks + kofA(t, lane)];
    const __bf16* bp = Wp + ks * 4 * 512 + lane * 16;
#pragma unroll
    for (int nt = 0; nt < 4; ++nt) {
      v16bf b = *(const v16bf*)(bp + nt * 512);
      acc[nt] = wmma_bf16(a, b, acc[nt]);
    }
  }
  int hi = (lane & 16) ? 8 : 0;
#pragma unroll
  for (int nt = 0; nt < 4; ++nt) {
    int n = nt * 16 + m;
    float bb = b_in[n];
#pragma unroll
    for (int r = 0; r < 8; ++r)
      h[(row0 + r + hi) * 64 + n] = acc[nt][r] + bb;
  }
}

// ---------------- edge scatter: agg[dst] += relu(h[src]) ----------------
__global__ __launch_bounds__(256) void scatter_kernel(
    const float* __restrict__ h, const int* __restrict__ ei,
    float* __restrict__ agg) {
  unsigned t = blockIdx.x * 256u + threadIdx.x;   // E*16 threads, 4 chans each
  int e = t >> 4;
  int c4 = (t & 15) << 2;
  int s = ei[e];
  int d = ei[Ee + e];
  float4 v = *(const float4*)(h + (long long)s * 64 + c4);
  v.x = v.x > 0.f ? v.x : 0.f;
  v.y = v.y > 0.f ? v.y : 0.f;
  v.z = v.z > 0.f ? v.z : 0.f;
  v.w = v.w > 0.f ? v.w : 0.f;
  float* ap = agg + (long long)d * 64 + c4;
  __hip_atomic_fetch_add(ap + 0, v.x, __ATOMIC_RELAXED, __HIP_MEMORY_SCOPE_AGENT);
  __hip_atomic_fetch_add(ap + 1, v.y, __ATOMIC_RELAXED, __HIP_MEMORY_SCOPE_AGENT);
  __hip_atomic_fetch_add(ap + 2, v.z, __ATOMIC_RELAXED, __HIP_MEMORY_SCOPE_AGENT);
  __hip_atomic_fetch_add(ap + 3, v.w, __ATOMIC_RELAXED, __HIP_MEMORY_SCOPE_AGENT);
}

// ---- fused GIN MLP: h += W2 @ relu(LN(((1+eps)h + agg) @ W1 + b1)) + b2 ----
__global__ __launch_bounds__(256) void gin_mlp_kernel(
    float* __restrict__ h, const float* __restrict__ agg,
    const __bf16* __restrict__ W1p, const __bf16* __restrict__ W2p,
    const float* __restrict__ b1, const float* __restrict__ gam,
    const float* __restrict__ bet, const float* __restrict__ b2,
    const float* __restrict__ eps_p) {
  __shared__ float Tb[8][16 * 128];
  int lane = threadIdx.x & 31;
  int wave = threadIdx.x >> 5;
  int tile = blockIdx.x * 8 + wave;
  if (tile * 16 >= Nn) return;
  long long row0 = (long long)tile * 16;
  float* T = &Tb[wave][0];
  float epsl = 1.0f + eps_p[0];

  // stage u = (1+eps)*h + agg  as [16][64]
  for (int i = lane; i < 16 * 64; i += 32)
    T[i] = epsl * h[row0 * 64 + i] + agg[row0 * 64 + i];
  lds_fence();

  v8f z8 = {0, 0, 0, 0, 0, 0, 0, 0};
  int m = lane & 15;
  int hi = (lane & 16) ? 8 : 0;

  // GEMM1: [16,64] @ [64,128]
  v8f acc[8];
#pragma unroll
  for (int i = 0; i < 8; ++i) acc[i] = z8;
#pragma unroll
  for (int ks = 0; ks < 2; ++ks) {
    v16bf a;
#pragma unroll
    for (int t = 0; t < 16; ++t) a[t] = (__bf16)T[m * 64 + 32 * ks + kofA(t, lane)];
    const __bf16* bp = W1p + ks * 8 * 512 + lane * 16;
#pragma unroll
    for (int nt = 0; nt < 8; ++nt) {
      v16bf b = *(const v16bf*)(bp + nt * 512);
      acc[nt] = wmma_bf16(a, b, acc[nt]);
    }
  }
  // store t1 = GEMM1 + b1 into LDS [16][128]
#pragma unroll
  for (int nt = 0; nt < 8; ++nt) {
    int n = nt * 16 + m;
    float bb = b1[n];
#pragma unroll
    for (int r = 0; r < 8; ++r) T[(r + hi) * 128 + n] = acc[nt][r] + bb;
  }
  lds_fence();

  // LayerNorm over 128 + relu: lane pair (l, l+16) handles row l halves
  int half = (lane >> 4) & 1;
  float s = 0.f, ss = 0.f;
  const float* Tr = T + m * 128 + half * 64;
#pragma unroll 8
  for (int c = 0; c < 64; ++c) { float v = Tr[c]; s += v; ss += v * v; }
  s += __shfl_xor(s, 16);
  ss += __shfl_xor(ss, 16);
  float mu = s * 0.0078125f;
  float var = ss * 0.0078125f - mu * mu;
  float rstd = rsqrtf(var + 1e-5f);
#pragma unroll 8
  for (int c = 0; c < 64; ++c) {
    int n = half * 64 + c;
    float v = (T[m * 128 + n] - mu) * rstd * gam[n] + bet[n];
    T[m * 128 + n] = v > 0.f ? v : 0.f;
  }
  lds_fence();

  // GEMM2: [16,128] @ [128,64], residual into h
  v8f acc2[4];
#pragma unroll
  for (int i = 0; i < 4; ++i) acc2[i] = z8;
#pragma unroll
  for (int ks = 0; ks < 4; ++ks) {
    v16bf a;
#pragma unroll
    for (int t = 0; t < 16; ++t) a[t] = (__bf16)T[m * 128 + 32 * ks + kofA(t, lane)];
    const __bf16* bp = W2p + ks * 4 * 512 + lane * 16;
#pragma unroll
    for (int nt = 0; nt < 4; ++nt) {
      v16bf b = *(const v16bf*)(bp + nt * 512);
      acc2[nt] = wmma_bf16(a, b, acc2[nt]);
    }
  }
#pragma unroll
  for (int nt = 0; nt < 4; ++nt) {
    int n = nt * 16 + m;
    float bb = b2[n];
#pragma unroll
    for (int r = 0; r < 8; ++r) {
      long long g = (row0 + r + hi) * 64 + n;
      h[g] += acc2[nt][r] + bb;
    }
  }
}

// ---------------- segment-sum pooling (batch is sorted) ----------------
__global__ __launch_bounds__(256) void pool_kernel(
    const float* __restrict__ h, const int* __restrict__ batch,
    float* __restrict__ pooled) {
  const int CH = 512;
  int c = threadIdx.x & 63;
  int sub = threadIdx.x >> 6;   // 0..3
  int start = blockIdx.x * CH;
  int end = start + CH;
  if (end > Nn) end = Nn;
  float acc = 0.f;
  int cur = -1;
  for (int nidx = start + sub; nidx < end; nidx += 4) {
    int g = batch[nidx];
    if (g != cur) {
      if (cur >= 0)
        __hip_atomic_fetch_add(&pooled[cur * 64 + c], acc, __ATOMIC_RELAXED,
                               __HIP_MEMORY_SCOPE_AGENT);
      acc = 0.f;
      cur = g;
    }
    acc += h[(long long)nidx * 64 + c];
  }
  if (cur >= 0)
    __hip_atomic_fetch_add(&pooled[cur * 64 + c], acc, __ATOMIC_RELAXED,
                           __HIP_MEMORY_SCOPE_AGENT);
}

// ---------------- out = relu(pooled@Wo1+bo1)@Wo2+bo2 ----------------
__global__ __launch_bounds__(64) void out_kernel(
    const float* __restrict__ pooled, const float* __restrict__ Wo1,
    const float* __restrict__ bo1, const float* __restrict__ Wo2,
    const float* __restrict__ bo2, float* __restrict__ out) {
  int g = threadIdx.x;
  if (g >= Gg) return;
  float p[64];
#pragma unroll
  for (int d = 0; d < 64; ++d) p[d] = pooled[g * 64 + d];
  float o = bo2[0];
  for (int hh = 0; hh < 128; ++hh) {
    float v = bo1[hh];
    for (int d = 0; d < 64; ++d) v = fmaf(p[d], Wo1[d * 128 + hh], v);
    o = fmaf(v > 0.f ? v : 0.f, Wo2[hh], o);
  }
  out[g] = o;
}

extern "C" void kernel_launch(void* const* d_in, const int* in_sizes, int n_in,
                              void* d_out, int out_size, void* d_ws, size_t ws_size,
                              hipStream_t stream) {
  const float* x    = (const float*)d_in[0];
  const float* W_in = (const float*)d_in[1];
  const float* b_in = (const float*)d_in[2];
  const float* eps  = (const float*)d_in[3];
  const float* W1   = (const float*)d_in[4];
  const float* b1   = (const float*)d_in[5];
  const float* gam  = (const float*)d_in[6];
  const float* bet  = (const float*)d_in[7];
  const float* W2   = (const float*)d_in[8];
  const float* b2   = (const float*)d_in[9];
  const float* Wo1  = (const float*)d_in[10];
  const float* bo1  = (const float*)d_in[11];
  const float* Wo2  = (const float*)d_in[12];
  const float* bo2  = (const float*)d_in[13];
  const int*   ei   = (const int*)d_in[14];
  const int*   bat  = (const int*)d_in[15];
  float* out = (float*)d_out;

  char* ws = (char*)d_ws;
  float*  h      = (float*)ws;                                        // N*64 f32
  float*  agg    = (float*)(ws + (size_t)Nn * 64 * 4);                // N*64 f32
  float*  pooled = (float*)(ws + (size_t)Nn * 64 * 4 * 2);            // G*64 f32
  __bf16* wp     = (__bf16*)(ws + (size_t)Nn * 64 * 4 * 2 + Gg * 64 * 4);
  __bf16* wp1 = wp + 4096;
  __bf16* wp2 = wp + 4096 + Ll * 8192;

  const int tiles = Nn / 16;                 // 6250
  const int mlp_blocks = (tiles + 7) / 8;    // 782

  pack_weights_kernel<<<(Ll * 64 * 128 + 255) / 256, 256, 0, stream>>>(W_in, W1, W2, wp);
  lin_in_kernel<<<mlp_blocks, 256, 0, stream>>>(x, wp, b_in, h);

  for (int l = 0; l < Ll; ++l) {
    hipMemsetAsync(agg, 0, (size_t)Nn * 64 * 4, stream);
    scatter_kernel<<<(Ee * 16) / 256, 256, 0, stream>>>(h, ei, agg);
    gin_mlp_kernel<<<mlp_blocks, 256, 0, stream>>>(
        h, agg, wp1 + l * 8192, wp2 + l * 8192, b1 + l * 128, gam + l * 128,
        bet + l * 128, b2 + l * 64, eps + l);
  }

  hipMemsetAsync(pooled, 0, (size_t)Gg * 64 * 4, stream);
  pool_kernel<<<(Nn + 511) / 512, 256, 0, stream>>>(h, bat, pooled);
  out_kernel<<<1, 64, 0, stream>>>(pooled, Wo1, bo1, Wo2, bo2, out);
}
